// FlaxHouseholderSelfAttention_78443282694778
// MI455X (gfx1250) — matrix operand order repaired and checked by reading, not compiled, not run on hardware
//
#include <hip/hip_runtime.h>
#include <hip/hip_bf16.h>
#include <stdint.h>

#ifndef __has_builtin
#define __has_builtin(x) 0
#endif

#if defined(__HIP_DEVICE_COMPILE__) && __has_builtin(__builtin_amdgcn_tensor_load_to_lds) && __has_builtin(__builtin_amdgcn_s_wait_tensorcnt)
#define USE_TDM 1
#else
#define USE_TDM 0
#endif

typedef __attribute__((ext_vector_type(16))) __bf16 bf16x16;
typedef __attribute__((ext_vector_type(8)))  __bf16 bf16x8;
typedef __attribute__((ext_vector_type(8)))  float  f32x8;
typedef __attribute__((ext_vector_type(4)))  unsigned int u32x4;
typedef __attribute__((ext_vector_type(8)))  int i32x8;
typedef __attribute__((ext_vector_type(4)))  int i32x4;

#define NB  2
#define NT  2048
#define NE  1024
#define NH  16
#define ND  64
#define NBT (NB*NT)

// ---------------------------------------------------------------- helpers ---

__device__ __forceinline__ f32x8 wmma_bf16(bf16x16 a, bf16x16 b, f32x8 c) {
#if defined(__HIP_DEVICE_COMPILE__)
  return __builtin_amdgcn_wmma_f32_16x16x32_bf16(false, a, false, b, (short)0, c, false, false);
#else
  return c;
#endif
}

// A-fragment (16-bit, 16x32): lane = row (M), K split {kb..kb+7, 16+kb..16+kb+7}
__device__ __forceinline__ bf16x16 fragA(const __bf16* base, int ld, int lane) {
  int r  = lane & 15;
  int kb = (lane & 16) ? 8 : 0;
  const __bf16* p = base + (size_t)r * ld + kb;
  bf16x8 lo = *(const bf16x8*)(p);
  bf16x8 hi = *(const bf16x8*)(p + 16);
  return __builtin_shufflevector(lo, hi, 0,1,2,3,4,5,6,7,8,9,10,11,12,13,14,15);
}

// B-fragment (16-bit, 32x16): lane = col (N), K split {kb..kb+7, 16+kb..16+kb+7}.
// Source laid out row-per-column (i.e. the N-major transpose), K contiguous.
__device__ __forceinline__ bf16x16 fragB(const __bf16* base, int ld, int lane) {
  int n  = lane & 15;
  int kb = (lane & 16) ? 8 : 0;
  const __bf16* p = base + (size_t)n * ld + kb;
  bf16x8 lo = *(const bf16x8*)(p);
  bf16x8 hi = *(const bf16x8*)(p + 16);
  return __builtin_shufflevector(lo, hi, 0,1,2,3,4,5,6,7,8,9,10,11,12,13,14,15);
}

#if USE_TDM
// 2D TDM load: tile_h rows x tile_w bf16 elems, src row stride in elems.
// LDS dest padded: 64B data + 16B pad per row (pad_interval=16 DW, pad_amount=4 DW).
__device__ __forceinline__ void tdm_load_tile(unsigned lds_addr, const void* gsrc,
                                              unsigned tile_w, unsigned tile_h,
                                              unsigned stride_elems) {
  unsigned long long ga = (unsigned long long)(uintptr_t)gsrc;
  u32x4 g0;
  g0[0] = 1u;                                              // count=1 (valid user D#)
  g0[1] = lds_addr;                                        // lds_addr (bytes)
  g0[2] = (unsigned)ga;                                    // global_addr[31:0]
  g0[3] = (unsigned)((ga >> 32) & 0x01FFFFFFu) | (2u<<30); // global_addr[56:32] | type=2
  i32x8 g1;
  g1[0] = (int)((1u << 16) | (1u << 20) | (3u << 22) | (3u << 25)); // data_size=2B, pad_en, int=16DW, amt=4DW
  g1[1] = (int)((tile_w & 0xFFFFu) << 16);                 // tensor_dim0[15:0]
  g1[2] = (int)((tile_w >> 16) | ((tile_h & 0xFFFFu) << 16)); // dim0 hi | tensor_dim1 lo
  g1[3] = (int)((tile_h >> 16) | ((tile_w & 0xFFFFu) << 16)); // dim1 hi | tile_dim0
  g1[4] = (int)(tile_h & 0xFFFFu);                         // tile_dim1 (tile_dim2=0)
  g1[5] = (int)stride_elems;                               // tensor_dim0_stride lo32
  g1[6] = 0;
  g1[7] = 0;
  i32x4 z4 = {0,0,0,0};
#if __clang_major__ >= 23
  i32x8 z8 = {0,0,0,0,0,0,0,0};
  __builtin_amdgcn_tensor_load_to_lds(g0, g1, z4, z4, z8, 0);
#else
  __builtin_amdgcn_tensor_load_to_lds(g0, g1, z4, z4, 0);
#endif
}
#endif

// ---------------------------------------------------------------- kernels ---

__global__ __launch_bounds__(256) void k_cvt(const float* __restrict__ in,
                                             __bf16* __restrict__ out, int n) {
  int i = blockIdx.x * 256 + threadIdx.x;
  if (i < n) out[i] = (__bf16)in[i];
}

// W[K,N] f32 -> Wt[N,K] bf16 (so B operands load K-contiguous)
__global__ __launch_bounds__(256) void k_cvt_t(const float* __restrict__ W,
                                               __bf16* __restrict__ Wt, int K, int N) {
  __shared__ float tile[32][33];
  int tx = threadIdx.x & 31, ty = threadIdx.x >> 5;
  int kb = blockIdx.y * 32, nb = blockIdx.x * 32;
#pragma unroll
  for (int i = ty; i < 32; i += 8) tile[i][tx] = W[(size_t)(kb + i) * N + nb + tx];
  __syncthreads();
#pragma unroll
  for (int i = ty; i < 32; i += 8) Wt[(size_t)(nb + i) * K + kb + tx] = (__bf16)tile[tx][i];
}

// C[M,N] f32 = A[M,K]bf16 * Bt[N,K]bf16^T + bias.  Block tile 128x128, 8 waves (2x4),
// each wave 4 m-tiles x 2 n-tiles.  K staged 32-wide into LDS via TDM (80B row stride).
__global__ __launch_bounds__(256) void k_gemm(const __bf16* __restrict__ A,
                                              const __bf16* __restrict__ Bt,
                                              const float* __restrict__ bias,
                                              float* __restrict__ C,
                                              int M, int N, int K) {
  __shared__ __bf16 lA[128 * 40];
  __shared__ __bf16 lB[128 * 40];
  int tid = threadIdx.x, lane = tid & 31, w = tid >> 5;
  int wm = w >> 2, wn = w & 3;
  int bm = blockIdx.y * 128, bn = blockIdx.x * 128;

  f32x8 acc[4][2];
#pragma unroll
  for (int mt = 0; mt < 4; ++mt)
#pragma unroll
    for (int nt = 0; nt < 2; ++nt) {
      f32x8 z = {0.f,0.f,0.f,0.f,0.f,0.f,0.f,0.f};
      acc[mt][nt] = z;
    }

  for (int k0 = 0; k0 < K; k0 += 32) {
    __syncthreads();
#if USE_TDM
    if (tid < 32) {
      tdm_load_tile((unsigned)(uintptr_t)(void*)lA, A + (size_t)bm * K + k0, 32u, 128u, (unsigned)K);
      tdm_load_tile((unsigned)(uintptr_t)(void*)lB, Bt + (size_t)bn * K + k0, 32u, 128u, (unsigned)K);
      __builtin_amdgcn_s_wait_tensorcnt(0);
    }
#else
    for (int i = tid; i < 128 * 32; i += 256) {
      int rr = i >> 5, cc = i & 31;
      lA[rr * 40 + cc] = A[(size_t)(bm + rr) * K + k0 + cc];
      lB[rr * 40 + cc] = Bt[(size_t)(bn + rr) * K + k0 + cc];
    }
#endif
    __syncthreads();
    if (k0 + 32 < K) {
      __builtin_prefetch(A + (size_t)(bm + (tid & 127)) * K + k0 + 32, 0, 1);
      __builtin_prefetch(Bt + (size_t)(bn + (tid & 127)) * K + k0 + 32, 0, 1);
    }
    bf16x16 af[4], bfm[2];
#pragma unroll
    for (int mt = 0; mt < 4; ++mt) af[mt] = fragA(lA + (wm * 64 + mt * 16) * 40, 40, lane);
#pragma unroll
    for (int nt = 0; nt < 2; ++nt) bfm[nt] = fragB(lB + (wn * 32 + nt * 16) * 40, 40, lane);
#pragma unroll
    for (int mt = 0; mt < 4; ++mt)
#pragma unroll
      for (int nt = 0; nt < 2; ++nt)
        acc[mt][nt] = wmma_bf16(af[mt], bfm[nt], acc[mt][nt]);
  }

  int rhi = (lane & 16) ? 8 : 0;
#pragma unroll
  for (int mt = 0; mt < 4; ++mt) {
    int row0 = bm + wm * 64 + mt * 16 + rhi;
#pragma unroll
    for (int nt = 0; nt < 2; ++nt) {
      int col = bn + wn * 32 + nt * 16 + (lane & 15);
      float bv = bias[col];
#pragma unroll
      for (int r = 0; r < 8; ++r)
        C[(size_t)(row0 + r) * N + col] = acc[mt][nt][r] + bv;
    }
  }
}

// Fused RoPE + Householder (q,k) + layout change. One wave per (b,h,t) row of 64.
__global__ __launch_bounds__(256) void k_ropehh(const float* __restrict__ q,
                                                const float* __restrict__ k,
                                                const float* __restrict__ v,
                                                const int* __restrict__ pos,
                                                const float* __restrict__ refl,
                                                __bf16* __restrict__ qh,
                                                __bf16* __restrict__ kh,
                                                __bf16* __restrict__ vt) {
  int tid = threadIdx.x, lane = tid & 31, w = tid >> 5;
  int row = blockIdx.x * 8 + w;       // (b*NH+h)*NT + t
  int bh = row >> 11;
  int t  = row & (NT - 1);
  int b = bh >> 4, h = bh & 15;
  size_t bt = (size_t)b * NT + t;
  const float2* qr = (const float2*)(q + bt * NE + h * ND);
  const float2* kr = (const float2*)(k + bt * NE + h * ND);
  const float2* vr = (const float2*)(v + bt * NE + h * ND);
  float2 xq = qr[lane], xk = kr[lane], xv = vr[lane];

  float p = (float)pos[t];
  float inv = __expf(-((float)(2 * lane) * (1.0f / 64.0f)) * 9.210340371976184f); // ln(10000)
  float ang = p * inv, sn, cs;
  __sincosf(ang, &sn, &cs);
  float2 rq = { xq.x * cs - xq.y * sn, xq.x * sn + xq.y * cs };
  float2 rk = { xk.x * cs - xk.y * sn, xk.x * sn + xk.y * cs };

#pragma unroll
  for (int i = 0; i < 4; ++i) {
    float2 vv = ((const float2*)(refl + i * ND))[lane];
    float n2 = vv.x * vv.x + vv.y * vv.y;
#pragma unroll
    for (int d = 1; d < 32; d <<= 1) n2 += __shfl_xor(n2, d, 32);
    float invn = rsqrtf(n2 + 1e-6f);
    vv.x *= invn; vv.y *= invn;
    float pq = rq.x * vv.x + rq.y * vv.y;
    float pk = rk.x * vv.x + rk.y * vv.y;
#pragma unroll
    for (int d = 1; d < 32; d <<= 1) { pq += __shfl_xor(pq, d, 32); pk += __shfl_xor(pk, d, 32); }
    rq.x -= 2.f * pq * vv.x; rq.y -= 2.f * pq * vv.y;
    rk.x -= 2.f * pk * vv.x; rk.y -= 2.f * pk * vv.y;
  }

  size_t ob = ((size_t)bh * NT + t) * ND + 2 * lane;
  qh[ob] = (__bf16)rq.x; qh[ob + 1] = (__bf16)rq.y;
  kh[ob] = (__bf16)rk.x; kh[ob + 1] = (__bf16)rk.y;
  size_t vb = ((size_t)bh * ND + 2 * lane) * NT + t;     // V transposed: [bh][d][t]
  vt[vb] = (__bf16)xv.x; vt[vb + NT] = (__bf16)xv.y;
}

// Attention: block = (b,h, 16-token tile). 8 waves x 256-kv slices; score panel in VGPRs.
__global__ __launch_bounds__(256) void k_attn(const __bf16* __restrict__ Qh,
                                              const __bf16* __restrict__ Kh,
                                              const __bf16* __restrict__ Vt,
                                              __bf16* __restrict__ Oh) {
  __shared__ float  oacc[16 * 64];
  __shared__ float  wmaxs[8 * 16], wsums[8 * 16], gmax[16], gsum[16];
  __shared__ __bf16 pbuf[8][16 * 40];
  int tid = threadIdx.x, lane = tid & 31, w = tid >> 5;
  int bh = blockIdx.x >> 7;            // / (NT/16)
  int qt = blockIdx.x & 127;
  const __bf16* Q = Qh + (size_t)bh * NT * ND;
  const __bf16* K = Kh + (size_t)bh * NT * ND;
  const __bf16* V = Vt + (size_t)bh * ND * NT;

  for (int i = tid; i < 16 * 64; i += 256) oacc[i] = 0.f;

  bf16x16 q0 = fragA(Q + (size_t)qt * 16 * ND,      ND, lane);
  bf16x16 q1 = fragA(Q + (size_t)qt * 16 * ND + 32, ND, lane);

  f32x8 s[16];
#pragma unroll
  for (int kt = 0; kt < 16; ++kt) {
    int kv = w * 256 + kt * 16;
    bf16x16 b0 = fragB(K + (size_t)kv * ND,      ND, lane);
    bf16x16 b1 = fragB(K + (size_t)kv * ND + 32, ND, lane);
    f32x8 z = {0.f,0.f,0.f,0.f,0.f,0.f,0.f,0.f};
    z = wmma_bf16(q0, b0, z);
    s[kt] = wmma_bf16(q1, b1, z);
  }

  int rbase = (lane & 16) ? 8 : 0;
#pragma unroll
  for (int r = 0; r < 8; ++r) {
    float m = s[0][r];
#pragma unroll
    for (int kt = 1; kt < 16; ++kt) m = fmaxf(m, s[kt][r]);
#pragma unroll
    for (int d = 1; d < 16; d <<= 1) m = fmaxf(m, __shfl_xor(m, d, 32));
    if ((lane & 15) == 0) wmaxs[w * 16 + rbase + r] = m;
  }
  __syncthreads();
  if (tid < 16) {
    float m = wmaxs[tid];
    for (int i = 1; i < 8; ++i) m = fmaxf(m, wmaxs[i * 16 + tid]);
    gmax[tid] = m;
  }
  __syncthreads();

  const float scale = 0.125f;  // 1/sqrt(64)
#pragma unroll
  for (int r = 0; r < 8; ++r) {
    float m = gmax[rbase + r], sum = 0.f;
#pragma unroll
    for (int kt = 0; kt < 16; ++kt) {
      float pp = __expf(scale * (s[kt][r] - m));
      s[kt][r] = pp; sum += pp;
    }
#pragma unroll
    for (int d = 1; d < 16; d <<= 1) sum += __shfl_xor(sum, d, 32);
    if ((lane & 15) == 0) wsums[w * 16 + rbase + r] = sum;
  }
  __syncthreads();
  if (tid < 16) {
    float ss = 0.f;
    for (int i = 0; i < 8; ++i) ss += wsums[i * 16 + tid];
    gsum[tid] = ss;
  }
  __syncthreads();

  f32x8 o[4];
#pragma unroll
  for (int dt = 0; dt < 4; ++dt) { f32x8 z = {0.f,0.f,0.f,0.f,0.f,0.f,0.f,0.f}; o[dt] = z; }
  __bf16* pb = &pbuf[w][0];
#pragma unroll
  for (int kp = 0; kp < 8; ++kp) {
#pragma unroll
    for (int tt = 0; tt < 2; ++tt) {
      int col = (lane & 15) + tt * 16;
#pragma unroll
      for (int r = 0; r < 8; ++r)
        pb[(rbase + r) * 40 + col] = (__bf16)s[kp * 2 + tt][r];   // transpose via per-wave LDS
    }
    asm volatile("" ::: "memory");
    bf16x16 ap = fragA(pb, 40, lane);
    int kv = w * 256 + kp * 32;
#pragma unroll
    for (int dt = 0; dt < 4; ++dt) {
      bf16x16 bv = fragB(V + (size_t)(dt * 16) * NT + kv, NT, lane);
      o[dt] = wmma_bf16(ap, bv, o[dt]);
    }
  }

#pragma unroll
  for (int dt = 0; dt < 4; ++dt) {
    int col = dt * 16 + (lane & 15);
#pragma unroll
    for (int r = 0; r < 8; ++r)
      atomicAdd(&oacc[(rbase + r) * 64 + col], o[dt][r]);   // ds_add_f32 cross-wave reduce
  }
  __syncthreads();

  int b = bh >> 4, h = bh & 15;
  for (int i = tid; i < 16 * 64; i += 256) {
    int rr = i >> 6, cc = i & 63;
    float val = oacc[i] / gsum[rr];
    Oh[((size_t)b * NT + qt * 16 + rr) * NE + h * ND + cc] = (__bf16)val;
  }
}

// ----------------------------------------------------------------- launch ---

extern "C" void kernel_launch(void* const* d_in, const int* in_sizes, int n_in,
                              void* d_out, int out_size, void* d_ws, size_t ws_size,
                              hipStream_t stream) {
  const float* x   = (const float*)d_in[0];
  const int*   pos = (const int*)  d_in[1];
  const float* Wq  = (const float*)d_in[2];
  const float* bq  = (const float*)d_in[3];
  const float* Wk  = (const float*)d_in[4];
  const float* bk  = (const float*)d_in[5];
  const float* Wv  = (const float*)d_in[6];
  const float* bv  = (const float*)d_in[7];
  const float* Wo  = (const float*)d_in[8];
  const float* bo  = (const float*)d_in[9];
  const float* rf  = (const float*)d_in[10];

  char* ws = (char*)d_ws;
  const size_t MB = (size_t)1 << 20;
  __bf16* xh  = (__bf16*)(ws);             //  8 MB  x bf16 [BT,E]
  __bf16* wqh = (__bf16*)(ws +  8 * MB);   //  2 MB  Wq^T bf16 [N,K]
  __bf16* wkh = (__bf16*)(ws + 10 * MB);
  __bf16* wvh = (__bf16*)(ws + 12 * MB);
  __bf16* woh = (__bf16*)(ws + 14 * MB);
  float*  qf  = (float*) (ws + 16 * MB);   // 16 MB  q f32 [BT,E]
  float*  kf  = (float*) (ws + 32 * MB);
  float*  vf  = (float*) (ws + 48 * MB);
  __bf16* qh  = (__bf16*)(ws + 64 * MB);   //  8 MB  [B,H,T,D]
  __bf16* kh  = (__bf16*)(ws + 72 * MB);
  __bf16* vt  = (__bf16*)(ws + 80 * MB);   //  8 MB  [B,H,D,T]
  __bf16* at  = (__bf16*)(ws + 88 * MB);   //  8 MB  attn bf16 [BT,E]

  k_cvt<<<(NBT * NE) / 256, 256, 0, stream>>>(x, xh, NBT * NE);
  dim3 tg(NE / 32, NE / 32);
  k_cvt_t<<<tg, 256, 0, stream>>>(Wq, wqh, NE, NE);
  k_cvt_t<<<tg, 256, 0, stream>>>(Wk, wkh, NE, NE);
  k_cvt_t<<<tg, 256, 0, stream>>>(Wv, wvh, NE, NE);
  k_cvt_t<<<tg, 256, 0, stream>>>(Wo, woh, NE, NE);

  dim3 gg(NE / 128, NBT / 128);
  k_gemm<<<gg, 256, 0, stream>>>(xh, wqh, bq, qf, NBT, NE, NE);
  k_gemm<<<gg, 256, 0, stream>>>(xh, wkh, bk, kf, NBT, NE, NE);
  k_gemm<<<gg, 256, 0, stream>>>(xh, wvh, bv, vf, NBT, NE, NE);

  k_ropehh<<<(NB * NH * NT) / 8, 256, 0, stream>>>(qf, kf, vf, pos, rf, qh, kh, vt);

  k_attn<<<NB * NH * (NT / 16), 256, 0, stream>>>(qh, kh, vt, at);

  k_gemm<<<gg, 256, 0, stream>>>(at, woh, bo, (float*)d_out, NBT, NE, NE);
}
